// NystromBlock_90263032693449
// MI455X (gfx1250) — matrix operand order repaired and checked by reading, not compiled, not run on hardware
//
#include <hip/hip_runtime.h>
#include <hip/hip_bf16.h>

#define BATCH 4
#define NH 8
#define SEQ 4096
#define HD 64
#define NL 64          // landmarks
#define SEG (SEQ / NL) // 64
#define KS 5
#define NITER 6
#define NSPLIT 8                       // S-splits for the flash pass
#define CPP (SEQ / 64 / NSPLIT)        // 64-row chunks per split = 8

#define SCALE 0.35355339059327373f // 64^-0.25
#define LDB 80                     // bf16 LDS row stride: 160B = 16B multiple -> b128-able

typedef __attribute__((ext_vector_type(8)))  float  v8f;
typedef __attribute__((ext_vector_type(2)))  float  v2f;
typedef __attribute__((ext_vector_type(16))) __bf16 v16bf;

// ---------------- WMMA fragment helpers (contiguous k = 16*half + j in A and B) ------------
// bf16 A fragment: A[M][K] 16x32 tile at (row0, kb). lane = (r, half).
__device__ __forceinline__ v16bf frag_a_bf16(const __bf16* src, int ld, int row0, int kb,
                                             int r, int half) {
  v16bf a;
#pragma unroll
  for (int j = 0; j < 16; ++j) {
    a[j] = src[(row0 + r) * ld + kb + 16 * half + j];
  }
  return a;
}
// bf16 B fragment from B stored [K][N]
__device__ __forceinline__ v16bf frag_b_bf16(const __bf16* src, int ld, int col0, int kb,
                                             int r, int half) {
  v16bf b;
#pragma unroll
  for (int j = 0; j < 16; ++j) b[j] = src[(kb + 16 * half + j) * ld + col0 + r];
  return b;
}
// bf16 B fragment where B = T^T, T stored [N][K]  (transposed read: 32B contiguous)
__device__ __forceinline__ v16bf frag_bt_bf16(const __bf16* src, int ld, int col0, int kb,
                                              int r, int half) {
  v16bf b;
#pragma unroll
  for (int j = 0; j < 16; ++j) b[j] = src[(col0 + r) * ld + kb + 16 * half + j];
  return b;
}

// 64x64x64 f32 matmul, wave computes its 2 tiles of the 4x4 tile grid.
// A stored [64][64] stride 64 in LDS; B supplied through functor bload(k, n).
template <typename BF>
__device__ __forceinline__ void mm64_f32(const float* A, BF bload, v8f acc[2],
                                         int r, int half, int wv) {
#pragma unroll
  for (int tt = 0; tt < 2; ++tt) {
    int t = wv * 2 + tt, tm = (t >> 2) * 16, tn = (t & 3) * 16;
    v8f c = {};
    for (int kk = 0; kk < NL; kk += 4) {
      int k0 = kk + 2 * half;
      v2f a, b;
      a[0] = A[(tm + r) * NL + k0];
      a[1] = A[(tm + r) * NL + k0 + 1];
      b[0] = bload(k0, tn + r);
      b[1] = bload(k0 + 1, tn + r);
      c = __builtin_amdgcn_wmma_f32_16x16x4_f32(false, a, false, b, (short)0, c, false, false);
    }
    acc[tt] = c;
  }
}

__device__ __forceinline__ void store64(float* D, const v8f acc[2], int r, int half, int wv,
                                        float mul) {
#pragma unroll
  for (int tt = 0; tt < 2; ++tt) {
    int t = wv * 2 + tt, tm = (t >> 2) * 16, tn = (t & 3) * 16;
#pragma unroll
    for (int v = 0; v < 8; ++v) D[(tm + 8 * half + v) * NL + tn + r] = mul * acc[tt][v];
  }
}

// ---------------- K1: landmark pooling -------------------------------------------------------
__global__ void __launch_bounds__(64) k1_pool(const float* __restrict__ Q,
                                              const float* __restrict__ K,
                                              float* __restrict__ qlf, float* __restrict__ klf,
                                              __bf16* __restrict__ qlb, __bf16* __restrict__ klb) {
  int lm = blockIdx.x;
  int bh = blockIdx.y;
  int d = threadIdx.x;
  const float* qp = Q + ((size_t)bh * SEQ + (size_t)lm * SEG) * HD + d;
  const float* kp = K + ((size_t)bh * SEQ + (size_t)lm * SEG) * HD + d;
  float qs = 0.f, ks = 0.f;
  for (int s = 0; s < SEG; ++s) { qs += qp[(size_t)s * HD]; ks += kp[(size_t)s * HD]; }
  float inv = SCALE / (float)SEG;
  qs *= inv; ks *= inv;
  size_t o = (size_t)bh * NL * HD + (size_t)lm * HD + d;
  qlf[o] = qs; klf[o] = ks;
  qlb[o] = (__bf16)qs; klb[o] = (__bf16)ks;
}

// ---------------- K2: kernel_2 softmax + Newton-Schulz inverse (all f32 WMMA) ----------------
__global__ void __launch_bounds__(256) k2_newton(const float* __restrict__ qlf,
                                                 const float* __restrict__ klf,
                                                 float* __restrict__ winv) {
  __shared__ float Km[NL * NL]; // kernel_2 (softmaxed), constant over iterations
  __shared__ float Vm[NL * NL]; // current inverse estimate
  __shared__ float Ab[NL * NL]; // scratch: ql, then KV
  __shared__ float Bb[NL * NL]; // scratch: kl, then M1/M2
  int bh = blockIdx.x, tid = threadIdx.x;
  int lane = tid & 31, r = lane & 15, half = lane >> 4, wv = tid >> 5;
  size_t base = (size_t)bh * NL * HD;

  for (int e = tid; e < NL * HD; e += 256) { Ab[e] = qlf[base + e]; Bb[e] = klf[base + e]; }
  __syncthreads();

  // scores = q_l @ k_l^T
  {
    v8f acc[2];
    mm64_f32(Ab, [&](int k, int n) { return Bb[n * NL + k]; }, acc, r, half, wv);
    store64(Km, acc, r, half, wv, 1.0f);
  }
  __syncthreads();

  // row softmax of Km (in place)
  if (tid < NL) {
    float mx = -1e30f;
    for (int j = 0; j < NL; ++j) mx = fmaxf(mx, Km[tid * NL + j]);
    float s = 0.f;
    for (int j = 0; j < NL; ++j) { float p = __expf(Km[tid * NL + j] - mx); Km[tid * NL + j] = p; s += p; }
    float inv = 1.f / s;
    for (int j = 0; j < NL; ++j) Km[tid * NL + j] *= inv;
  }
  __syncthreads();

  // init = max over columns of column-sums; alpha = 1/init
  if (tid < NL) { float s = 0.f; for (int i = 0; i < NL; ++i) s += Km[i * NL + tid]; Vm[tid] = s; }
  __syncthreads();
  if (tid == 0) { float mx = Vm[0]; for (int j = 1; j < NL; ++j) mx = fmaxf(mx, Vm[j]); Vm[NL] = 1.f / mx; }
  __syncthreads();
  float alpha = Vm[NL];
  __syncthreads();
  for (int e = tid; e < NL * NL; e += 256) { int i = e >> 6, j = e & 63; Vm[i * NL + j] = alpha * Km[j * NL + i]; }
  __syncthreads();

  // Newton-Schulz: Vm = 0.25*Vm @ (13I - KV@(15I - KV@(7I - KV))),  KV = Km@Vm
  for (int it = 0; it < NITER; ++it) {
    v8f acc[2];
    // KV -> Ab (Ab is dead here)
    mm64_f32(Km, [&](int k, int n) { return Vm[k * NL + n]; }, acc, r, half, wv);
    store64(Ab, acc, r, half, wv, 1.0f);
    __syncthreads();
    // M1 = KV @ (7I - KV) -> Bb (Bb is dead here)
    mm64_f32(Ab, [&](int k, int n) { return ((k == n) ? 7.f : 0.f) - Ab[k * NL + n]; }, acc, r, half, wv);
    store64(Bb, acc, r, half, wv, 1.0f);
    __syncthreads();
    // M2 = KV @ (15I - M1) -> Bb (overwrites input; barrier between compute and store)
    mm64_f32(Ab, [&](int k, int n) { return ((k == n) ? 15.f : 0.f) - Bb[k * NL + n]; }, acc, r, half, wv);
    __syncthreads();
    store64(Bb, acc, r, half, wv, 1.0f);
    __syncthreads();
    // Vm = 0.25 * Vm @ (13I - M2)   (overwrites Vm; barrier between compute and store)
    mm64_f32(Vm, [&](int k, int n) { return ((k == n) ? 13.f : 0.f) - Bb[k * NL + n]; }, acc, r, half, wv);
    __syncthreads();
    store64(Vm, acc, r, half, wv, 0.25f);
    __syncthreads();
  }
  for (int e = tid; e < NL * NL; e += 256) winv[(size_t)bh * NL * NL + e] = Vm[e];
}

// ---------------- K3a: partial flash  F_p = sum exp(q_l k^T - m_p) V  over one S-split -------
__global__ void __launch_bounds__(256) k3_flash_part(const float* __restrict__ Kg,
                                                     const float* __restrict__ Vg,
                                                     const __bf16* __restrict__ qlb,
                                                     float* __restrict__ partF,
                                                     float* __restrict__ partM,
                                                     float* __restrict__ partL) {
  __shared__ __attribute__((aligned(16))) __bf16 qls[NL * LDB];
  __shared__ __attribute__((aligned(16))) __bf16 Kb[64 * LDB];
  __shared__ __attribute__((aligned(16))) __bf16 Vb[64 * LDB];
  __shared__ __attribute__((aligned(16))) __bf16 Pb[NL * LDB];
  __shared__ float  Sc[NL * 65];
  __shared__ float  rowM[NL], rowL[NL], fac[NL];
  int bh = blockIdx.x, part = blockIdx.y, tid = threadIdx.x;
  int lane = tid & 31, r = lane & 15, half = lane >> 4, wv = tid >> 5;
  size_t sbase = (size_t)bh * SEQ * HD;
  size_t lbase = (size_t)bh * NL * HD;

  for (int e = tid; e < NL * HD; e += 256) { int i = e >> 6, j = e & 63; qls[i * LDB + j] = qlb[lbase + e]; }
  if (tid < NL) { rowM[tid] = -1e30f; rowL[tid] = 0.f; }
  v8f acc[2] = {v8f{}, v8f{}};
  __syncthreads();

  for (int c = 0; c < CPP; ++c) {
    int s0 = (part * CPP + c) * 64;
    for (int e = tid; e < 64 * HD; e += 256) {
      int i = e >> 6, j = e & 63;
      Kb[i * LDB + j] = (__bf16)(Kg[sbase + (size_t)(s0 + i) * HD + j] * SCALE);
      Vb[i * LDB + j] = (__bf16)(Vg[sbase + (size_t)(s0 + i) * HD + j]);
    }
    // prefetch next chunk while this one is consumed (global_prefetch_b8)
    if (c + 1 < CPP) {
      size_t nb = sbase + (size_t)(s0 + 64) * HD + (size_t)tid * 16;
      __builtin_prefetch(Kg + nb, 0, 1);
      __builtin_prefetch(Vg + nb, 0, 1);
    }
    __syncthreads();
    // scores = q_l @ K_chunk^T   [m x 64], K-dim = D = 64
#pragma unroll
    for (int tt = 0; tt < 2; ++tt) {
      int t = wv * 2 + tt, tm = (t >> 2) * 16, tn = (t & 3) * 16;
      v8f s = {};
      for (int kb = 0; kb < HD; kb += 32) {
        v16bf a = frag_a_bf16(qls, LDB, tm, kb, r, half);
        v16bf b = frag_bt_bf16(Kb, LDB, tn, kb, r, half);
        s = __builtin_amdgcn_wmma_f32_16x16x32_bf16(false, a, false, b, (short)0, s, false, false);
      }
#pragma unroll
      for (int v = 0; v < 8; ++v) Sc[(tm + 8 * half + v) * 65 + tn + r] = s[v];
    }
    __syncthreads();
    // online softmax update per landmark row
    if (tid < NL) {
      float mold = rowM[tid], mx = mold;
      for (int j = 0; j < 64; ++j) mx = fmaxf(mx, Sc[tid * 65 + j]);
      float f = __expf(mold - mx);
      float l = rowL[tid] * f;
      for (int j = 0; j < 64; ++j) {
        float p = __expf(Sc[tid * 65 + j] - mx);
        Pb[tid * LDB + j] = (__bf16)p;
        l += p;
      }
      rowM[tid] = mx; rowL[tid] = l; fac[tid] = f;
    }
    __syncthreads();
    // acc = acc*fac + P @ V_chunk
#pragma unroll
    for (int tt = 0; tt < 2; ++tt) {
      int t = wv * 2 + tt, tm = (t >> 2) * 16, tn = (t & 3) * 16;
      v8f c2 = acc[tt];
#pragma unroll
      for (int v = 0; v < 8; ++v) c2[v] *= fac[tm + 8 * half + v];
      for (int kb = 0; kb < 64; kb += 32) {
        v16bf a = frag_a_bf16(Pb, LDB, tm, kb, r, half);
        v16bf b = frag_b_bf16(Vb, LDB, tn, kb, r, half);
        c2 = __builtin_amdgcn_wmma_f32_16x16x32_bf16(false, a, false, b, (short)0, c2, false, false);
      }
      acc[tt] = c2;
    }
    __syncthreads();
  }
  // write unnormalized partial state
  size_t pbase = ((size_t)bh * NSPLIT + part) * (NL * HD);
#pragma unroll
  for (int tt = 0; tt < 2; ++tt) {
    int t = wv * 2 + tt, tm = (t >> 2) * 16, tn = (t & 3) * 16;
#pragma unroll
    for (int v = 0; v < 8; ++v) {
      int row = tm + 8 * half + v;
      partF[pbase + (size_t)row * HD + tn + r] = acc[tt][v];
    }
  }
  if (tid < NL) {
    partM[((size_t)bh * NSPLIT + part) * NL + tid] = rowM[tid];
    partL[((size_t)bh * NSPLIT + part) * NL + tid] = rowL[tid];
  }
}

// ---------------- K3b: combine S-split partials ----------------------------------------------
__global__ void __launch_bounds__(256) k3_reduce(const float* __restrict__ partF,
                                                 const float* __restrict__ partM,
                                                 const float* __restrict__ partL,
                                                 float* __restrict__ Fm) {
  __shared__ float wgt[NSPLIT][NL];
  __shared__ float invL[NL];
  int bh = blockIdx.x, tid = threadIdx.x;
  if (tid < NL) {
    float M = -1e30f;
    for (int p = 0; p < NSPLIT; ++p)
      M = fmaxf(M, partM[((size_t)bh * NSPLIT + p) * NL + tid]);
    float L = 0.f;
    for (int p = 0; p < NSPLIT; ++p) {
      float w = __expf(partM[((size_t)bh * NSPLIT + p) * NL + tid] - M);
      wgt[p][tid] = w;
      L += w * partL[((size_t)bh * NSPLIT + p) * NL + tid];
    }
    invL[tid] = 1.f / L;
  }
  __syncthreads();
  for (int e = tid; e < NL * HD; e += 256) {
    int row = e >> 6;
    float f = 0.f;
#pragma unroll
    for (int p = 0; p < NSPLIT; ++p)
      f += wgt[p][row] * partF[((size_t)bh * NSPLIT + p) * (NL * HD) + e];
    Fm[(size_t)bh * NL * HD + e] = f * invL[row];
  }
}

// ---------------- K4: G = W^-1 @ F  (f32 WMMA), stored bf16 ----------------------------------
__global__ void __launch_bounds__(256) k4_mix(const float* __restrict__ winv,
                                              const float* __restrict__ Fm,
                                              __bf16* __restrict__ Gb) {
  __shared__ float Wm[NL * 65];
  __shared__ float Fs[NL * 65];
  int bh = blockIdx.x, tid = threadIdx.x;
  int lane = tid & 31, r = lane & 15, half = lane >> 4, wv = tid >> 5;
  size_t base = (size_t)bh * NL * HD;
  for (int e = tid; e < NL * HD; e += 256) {
    int i = e >> 6, j = e & 63;
    Wm[i * 65 + j] = winv[base + e];
    Fs[i * 65 + j] = Fm[base + e];
  }
  __syncthreads();
#pragma unroll
  for (int tt = 0; tt < 2; ++tt) {
    int t = wv * 2 + tt, tm = (t >> 2) * 16, tn = (t & 3) * 16;
    v8f acc = {};
    for (int kk = 0; kk < NL; kk += 4) {
      int k0 = kk + 2 * half;
      v2f a, b;
      a[0] = Wm[(tm + r) * 65 + k0];
      a[1] = Wm[(tm + r) * 65 + k0 + 1];
      b[0] = Fs[k0 * 65 + tn + r];
      b[1] = Fs[(k0 + 1) * 65 + tn + r];
      acc = __builtin_amdgcn_wmma_f32_16x16x4_f32(false, a, false, b, (short)0, acc, false, false);
    }
#pragma unroll
    for (int v = 0; v < 8; ++v)
      Gb[base + (size_t)(tm + 8 * half + v) * HD + tn + r] = (__bf16)acc[v];
  }
}

// ---------------- K5: X = softmax(q k_l^T) @ G  + depthwise conv(V) --------------------------
__global__ void __launch_bounds__(256) k5_out(const float* __restrict__ Qg,
                                              const float* __restrict__ Vg,
                                              const float* __restrict__ convw,
                                              const __bf16* __restrict__ klb,
                                              const __bf16* __restrict__ Gb,
                                              float* __restrict__ out) {
  __shared__ __attribute__((aligned(16))) __bf16 qb[64 * LDB];
  __shared__ __attribute__((aligned(16))) __bf16 kls[NL * LDB];
  __shared__ __attribute__((aligned(16))) __bf16 Gs[NL * LDB];
  __shared__ __attribute__((aligned(16))) __bf16 Pb[64 * LDB];
  __shared__ float  uni[68 * 64]; // first: scores (stride 65); later: V tile rows s0-2..s0+65
  __shared__ float  wsh[KS];
  int s0 = blockIdx.x * 64;
  int bh = blockIdx.y;
  int h = bh % NH;
  int tid = threadIdx.x;
  int lane = tid & 31, r = lane & 15, half = lane >> 4, wv = tid >> 5;
  size_t sbase = (size_t)bh * SEQ * HD;
  size_t lbase = (size_t)bh * NL * HD;

  for (int e = tid; e < 64 * HD; e += 256) {
    int i = e >> 6, j = e & 63;
    qb[i * LDB + j] = (__bf16)(Qg[sbase + (size_t)(s0 + i) * HD + j] * SCALE);
  }
  for (int e = tid; e < NL * HD; e += 256) {
    int i = e >> 6, j = e & 63;
    kls[i * LDB + j] = klb[lbase + e];
    Gs[i * LDB + j]  = Gb[lbase + e];
  }
  if (tid < KS) wsh[tid] = convw[h * KS + tid];
  __syncthreads();

  // scores = q_tile @ k_l^T
  float* Sc = uni; // stride 65
#pragma unroll
  for (int tt = 0; tt < 2; ++tt) {
    int t = wv * 2 + tt, tm = (t >> 2) * 16, tn = (t & 3) * 16;
    v8f s = {};
    for (int kb = 0; kb < HD; kb += 32) {
      v16bf a = frag_a_bf16(qb, LDB, tm, kb, r, half);
      v16bf b = frag_bt_bf16(kls, LDB, tn, kb, r, half);
      s = __builtin_amdgcn_wmma_f32_16x16x32_bf16(false, a, false, b, (short)0, s, false, false);
    }
#pragma unroll
    for (int v = 0; v < 8; ++v) Sc[(tm + 8 * half + v) * 65 + tn + r] = s[v];
  }
  __syncthreads();

  // full softmax over m=64 per sequence row
  if (tid < 64) {
    float mx = -1e30f;
    for (int j = 0; j < NL; ++j) mx = fmaxf(mx, Sc[tid * 65 + j]);
    float sum = 0.f;
    for (int j = 0; j < NL; ++j) { float p = __expf(Sc[tid * 65 + j] - mx); Sc[tid * 65 + j] = p; sum += p; }
    float inv = 1.f / sum;
    for (int j = 0; j < NL; ++j) Pb[tid * LDB + j] = (__bf16)(Sc[tid * 65 + j] * inv);
  }
  __syncthreads();

  // scores buffer is dead now: reuse as the V tile for the conv while WMMA runs
  float* Vt = uni; // stride 64, 68 rows
  for (int e = tid; e < 68 * HD; e += 256) {
    int i = e >> 6, j = e & 63;
    int s = s0 - 2 + i;
    Vt[i * HD + j] = (s >= 0 && s < SEQ) ? Vg[sbase + (size_t)s * HD + j] : 0.f;
  }
  v8f acc[2];
#pragma unroll
  for (int tt = 0; tt < 2; ++tt) {
    int t = wv * 2 + tt, tm = (t >> 2) * 16, tn = (t & 3) * 16;
    v8f c = {};
    for (int kb = 0; kb < NL; kb += 32) {
      v16bf a = frag_a_bf16(Pb, LDB, tm, kb, r, half);
      v16bf b = frag_b_bf16(Gs, LDB, tn, kb, r, half);
      c = __builtin_amdgcn_wmma_f32_16x16x32_bf16(false, a, false, b, (short)0, c, false, false);
    }
    acc[tt] = c;
  }
  __syncthreads();

#pragma unroll
  for (int tt = 0; tt < 2; ++tt) {
    int t = wv * 2 + tt, tm = (t >> 2) * 16, tn = (t & 3) * 16;
#pragma unroll
    for (int v = 0; v < 8; ++v) {
      int row = tm + 8 * half + v;
      int n = tn + r;
      float cv = 0.f;
#pragma unroll
      for (int q = 0; q < KS; ++q) cv += wsh[q] * Vt[(row + q) * HD + n];
      out[sbase + (size_t)(s0 + row) * HD + n] = acc[tt][v] + cv;
    }
  }
}

// ---------------- launcher -------------------------------------------------------------------
extern "C" void kernel_launch(void* const* d_in, const int* in_sizes, int n_in,
                              void* d_out, int out_size, void* d_ws, size_t ws_size,
                              hipStream_t stream) {
  (void)in_sizes; (void)n_in; (void)out_size; (void)ws_size;
  const float* Q      = (const float*)d_in[0];
  const float* K      = (const float*)d_in[1];
  const float* V      = (const float*)d_in[2];
  const float* conv_w = (const float*)d_in[3];
  float* out = (float*)d_out;

  char* ws = (char*)d_ws;
  const size_t NM = (size_t)BATCH * NH * NL * HD; // 131072 elems per per-head 64x64 matrix set
  float*  qlf   = (float*)ws;  ws += NM * sizeof(float);
  float*  klf   = (float*)ws;  ws += NM * sizeof(float);
  float*  winv  = (float*)ws;  ws += NM * sizeof(float);
  float*  Fm    = (float*)ws;  ws += NM * sizeof(float);
  float*  partF = (float*)ws;  ws += NM * NSPLIT * sizeof(float);          // 4 MB
  float*  partM = (float*)ws;  ws += (size_t)BATCH * NH * NSPLIT * NL * sizeof(float);
  float*  partL = (float*)ws;  ws += (size_t)BATCH * NH * NSPLIT * NL * sizeof(float);
  __bf16* qlb   = (__bf16*)ws; ws += NM * sizeof(__bf16);
  __bf16* klb   = (__bf16*)ws; ws += NM * sizeof(__bf16);
  __bf16* Gb    = (__bf16*)ws; ws += NM * sizeof(__bf16);
  // total workspace use: ~7 MB

  k1_pool      <<<dim3(NL, BATCH * NH),      64,  0, stream>>>(Q, K, qlf, klf, qlb, klb);
  k2_newton    <<<dim3(BATCH * NH),          256, 0, stream>>>(qlf, klf, winv);
  k3_flash_part<<<dim3(BATCH * NH, NSPLIT),  256, 0, stream>>>(K, V, qlb, partF, partM, partL);
  k3_reduce    <<<dim3(BATCH * NH),          256, 0, stream>>>(partF, partM, partL, Fm);
  k4_mix       <<<dim3(BATCH * NH),          256, 0, stream>>>(winv, Fm, Gb);
  k5_out       <<<dim3(SEQ / 64, BATCH * NH), 256, 0, stream>>>(Q, V, conv_w, klb, Gb, out);
}
